// HandNodeEncoder_60825326846771
// MI455X (gfx1250) — matrix operand order repaired and checked by reading, not compiled, not run on hardware
//
#include <hip/hip_runtime.h>
#include <hip/hip_fp16.h>

typedef __attribute__((ext_vector_type(16))) _Float16 v16h;
typedef __attribute__((ext_vector_type(8)))  _Float16 v8h;
typedef __attribute__((ext_vector_type(8)))  float    v8f;
typedef unsigned int u32x4 __attribute__((ext_vector_type(4)));
typedef unsigned int u32x8 __attribute__((ext_vector_type(8)));

#define DEV __device__ __forceinline__

constexpr int   NODES = 21;
constexpr int   DM    = 256;
constexpr int   NKEY  = 196;
constexpr int   LD16  = 264;   // f16 row stride (mult of 8 -> 16B aligned rows)
constexpr int   LDF   = 264;   // f32 row stride
constexpr int   LDS_S = 40;    // score row stride (40*2B = 80 = 16*5, aligned)
constexpr float SCALE = 0.17677669529663689f; // 1/sqrt(32)
constexpr float NEGF  = -1e30f;

// ---- dynamic shared memory layout (bytes); all 32-row buffers ----
constexpr size_t OFF_H16 = 0;
constexpr size_t OFF_Q16 = OFF_H16 + (size_t)32*LD16*2;
constexpr size_t OFF_X16 = OFF_Q16 + (size_t)32*LD16*2;
constexpr size_t OFF_KV  = OFF_X16 + (size_t)32*LD16*2;
constexpr size_t OFF_S16 = OFF_KV  + (size_t)32*LD16*2;    // scores; aliases GAT scratch
constexpr size_t OFF_TA  = OFF_S16 + (size_t)8*32*LDS_S*2; // TDM staging A / f32 scratch T
constexpr size_t OFF_TB  = OFF_TA  + (size_t)32*LDF*4;     // TDM staging B
constexpr size_t OFF_ST  = OFF_TB  + (size_t)32*LDF*4;
constexpr size_t SMEM_BYTES = OFF_ST + 800*4;              // ~158.8 KB

__device__ const int EDGES[23][2] = {
 {0,1},{1,2},{2,3},{3,4},{0,5},{5,6},{6,7},{7,8},{0,9},{9,10},{10,11},{11,12},
 {0,13},{13,14},{14,15},{15,16},{0,17},{17,18},{18,19},{19,20},{5,9},{9,13},{13,17}};

struct Params {
  const float *node_feats, *img;
  const float *mlp_w1, *mlp_b1, *mlp_w2, *mlp_b2;
  const float *gat_W, *gat_a1, *gat_a2, *gat_pw, *gat_pb, *gat_g, *gat_b;
  const float *ca_qw, *ca_qb, *ca_kw, *ca_kb, *ca_vw, *ca_vb, *ca_ow, *ca_ob;
  const float *ca_g1, *ca_b1, *ca_g2, *ca_b2, *ca_f1w, *ca_f1b, *ca_f2w, *ca_f2b;
  const float *tr_qw, *tr_qb, *tr_kw, *tr_kb, *tr_vw, *tr_vb, *tr_ow, *tr_ob;
  const float *tr_g1, *tr_b1, *tr_g2, *tr_b2, *tr_f1w, *tr_f1b, *tr_f2w, *tr_f2b;
  float* out;
};

DEV int laneid(){ return (int)(threadIdx.x & 31u); }

// ---- WMMA fragment helpers per CDNA5 ISA 7.12.2 (wave32) ----
DEV v16h frag_rowpair(const _Float16* base, int lda){
  const int la=laneid(); const int m=la&15; const int g=la>>4;
  const _Float16* r0 = base + (size_t)m*lda + g*8;
  v8h lo = *(const v8h*)(r0);
  v8h hi = *(const v8h*)(r0 + 16);
  return __builtin_shufflevector(lo, hi, 0,1,2,3,4,5,6,7,8,9,10,11,12,13,14,15);
}
DEV v16h frag_rowpair_f32(const float* base, int lda){
  const int la=laneid(); const int m=la&15; const int g=la>>4;
  const float* r0 = base + (size_t)m*lda + g*8;
  v16h a;
  #pragma unroll
  for(int i=0;i<8;++i) a[i]   = (_Float16)r0[i];
  #pragma unroll
  for(int i=0;i<8;++i) a[8+i] = (_Float16)r0[16+i];
  return a;
}
DEV v16h frag_bt(const _Float16* base, int ldb){
  const int la=laneid(); const int n=la&15; const int g=la>>4;
  const _Float16* r0 = base + (size_t)n*ldb + g*8;
  v8h lo = *(const v8h*)(r0);
  v8h hi = *(const v8h*)(r0 + 16);
  return __builtin_shufflevector(lo, hi, 0,1,2,3,4,5,6,7,8,9,10,11,12,13,14,15);
}
template<typename F> DEV v16h frag_b(F f){
  v16h bV; const int la=laneid(); const int n=la&15; const int g=la>>4;
  #pragma unroll
  for(int e=0;e<16;++e) bV[e] = f(g*8+e+(e&8), n);
  return bV;
}
template<typename F> DEV void frag_c_store(const v8f& c, F f){
  const int la=laneid(); const int n=la&15; const int g=la>>4;
  #pragma unroll
  for(int r=0;r<8;++r) f(r + g*8, n, c[r]);
}

DEV v8f wmma16(v16h a, v16h b, v8f c){
  return __builtin_amdgcn_wmma_f32_16x16x32_f16(false, a, false, b, (short)0, c, false, false);
}

DEV float wred(float v){
  #pragma unroll
  for(int o=16;o>0;o>>=1) v += __shfl_xor(v, o, 32);
  return v;
}

DEV float gelu_exact(float x){ return 0.5f*x*(1.f+erff(x*0.70710678118654752f)); }

DEV void lnorm(const float* T, const float* g, const float* bb, _Float16* Ho, float* gout){
  const int wid=(int)(threadIdx.x>>5), la=(int)(threadIdx.x&31u);
  for(int r=wid; r<NODES; r+=8){
    float s=0.f, s2=0.f;
    #pragma unroll
    for(int c=la; c<DM; c+=32){ float x=T[r*LDF+c]; s+=x; s2=fmaf(x,x,s2); }
    s=wred(s); s2=wred(s2);
    const float mu  = s*(1.f/DM);
    const float var = s2*(1.f/DM) - mu*mu;
    const float inv = rsqrtf(var + 1e-5f);
    #pragma unroll
    for(int c=la; c<DM; c+=32){
      float y = (T[r*LDF+c]-mu)*inv*g[c] + bb[c];
      Ho[r*LD16+c] = (_Float16)y;
      if(gout) gout[(size_t)r*DM + c] = y;
    }
  }
}

// C[32 x 256] = A[32 x K] * Bf ; 32 output tiles over 8 waves, no guards anywhere.
template<typename AL, typename BF, typename EPI>
DEV void gemm_core(int K, AL aload, BF bf, EPI epi){
  const int wid=(int)(threadIdx.x>>5);
  for(int t=wid; t<32; t+=8){
    const int mi=t&1, ni=t>>1;
    v8f acc = {};
    for(int k0=0;k0<K;k0+=32){
      v16h a  = aload(mi, k0);
      v16h bV = frag_b([&](int kk,int n)->_Float16{
        return (_Float16)bf(k0+kk, ni*16+n); });
      acc = wmma16(a, bV, acc);
    }
    frag_c_store(acc, [&](int m,int n,float v){ epi(mi*16+m, ni*16+n, v); });
  }
}
template<typename BF, typename EPI>
DEV void gemm16(const _Float16* A, int lda, int K, BF bf, EPI epi){
  gemm_core(K, [&](int mi,int k0){ return frag_rowpair(A + (size_t)(mi*16)*lda + k0, lda); }, bf, epi);
}
template<typename BF, typename EPI>
DEV void gemm_f32A(const float* A, int lda, int K, BF bf, EPI epi){
  gemm_core(K, [&](int mi,int k0){ return frag_rowpair_f32(A + (size_t)(mi*16)*lda + k0, lda); }, bf, epi);
}

// FFN: T init = f2b + residual(H); chunked hidden; exact GELU; accumulate; LN
DEV void ffn_block(const float* f1w, const float* f1b, const float* f2w, const float* f2b,
                   int hidden, const float* g, const float* bta,
                   _Float16* H16, _Float16* Q16, float* T, float* gout){
  const int tid=(int)threadIdx.x;
  for(int idx=tid; idx<NODES*DM; idx+=256){
    const int i=idx>>8, c=idx&255;
    T[i*LDF+c] = f2b[c] + (float)H16[i*LD16+c];
  }
  __syncthreads();
  for(int c0=0; c0<hidden; c0+=256){
    gemm16(H16, LD16, DM,
      [&](int k,int c){ return f1w[(size_t)k*hidden + c0 + c]; },
      [&](int m,int n,float v){ Q16[m*LD16+n] = (_Float16)gelu_exact(v + f1b[c0+n]); });
    __syncthreads();
    gemm16(Q16, LD16, DM,
      [&](int k,int c){ return f2w[(size_t)(c0+k)*DM + c]; },
      [&](int m,int n,float v){ T[m*LDF+n] += v; });
    __syncthreads();
  }
  lnorm(T, g, bta, H16, gout);
  __syncthreads();
}

__global__ __launch_bounds__(256,1) void hand_encoder_fused(Params p){
  extern __shared__ char smem[];
  _Float16* H16 = (_Float16*)(smem + OFF_H16);
  _Float16* Q16 = (_Float16*)(smem + OFF_Q16);
  _Float16* X16 = (_Float16*)(smem + OFF_X16);
  _Float16* KV16= (_Float16*)(smem + OFF_KV);   // K row-major / V^T [256][32]
  _Float16* S16 = (_Float16*)(smem + OFF_S16);  // [8][32][LDS_S]
  float*    TA  = (float*)   (smem + OFF_TA);   // TDM staging 0 / f32 scratch
  float*    TB  = (float*)   (smem + OFF_TB);   // TDM staging 1
  float*    T   = TA;                           // f32 scratch outside flash loops
  // GAT-only scratch aliased into the score region (disjoint in time)
  float* s_al  = (float*)(smem + OFF_S16);      // [4][21][21]
  float* s_s1  = s_al + NODES*NODES*4;          // [4][21] (padded)
  float* s_s2  = s_s1 + 96;
  float*    st  = (float*)   (smem + OFF_ST);
  float* s_m   = st;                 // [8][32]
  float* s_l   = st + 256;           // [8][32]
  float* s_cor = st + 512;           // [8][32]
  unsigned* s_adj = (unsigned*)(st + 768);

  const int b   = (int)blockIdx.x;
  const int tid = (int)threadIdx.x;
  const int wid = tid >> 5;

  if(tid < NODES){
    unsigned m = 1u << tid;
    for(int e=0;e<23;++e){
      if(EDGES[e][0]==tid) m |= 1u<<EDGES[e][1];
      if(EDGES[e][1]==tid) m |= 1u<<EDGES[e][0];
    }
    s_adj[tid]=m;
  }
  // X16 is the only buffer read as A before being fully written: zero its pad rows once.
  for(int idx=tid; idx<(32-NODES)*LD16; idx+=256) X16[NODES*LD16+idx] = (_Float16)0.f;

  // ================= node MLP =================
  for(int idx=tid; idx<NODES*DM; idx+=256){
    const int i=idx>>8, c=idx&255;
    const float* nf = p.node_feats + ((size_t)b*NODES + i)*26;
    float s = p.mlp_b1[c];
    #pragma unroll
    for(int k=0;k<26;++k) s = fmaf(nf[k], p.mlp_w1[k*DM+c], s);
    X16[i*LD16+c] = (_Float16)fmaxf(s, 0.f);
  }
  __syncthreads();
  gemm16(X16, LD16, DM,
    [&](int k,int c){ return p.mlp_w2[k*DM+c]; },
    [&](int m,int n,float v){ H16[m*LD16+n] = (_Float16)(v + p.mlp_b2[n]); });
  __syncthreads();

  // ================= GAT x2 =================
  for(int L=0; L<2; ++L){
    gemm16(H16, LD16, DM,                               // Wh (4 heads concat)
      [&](int k,int c){ const int h=c>>6, o=c&63;
        return p.gat_W[((size_t)(L*4+h)*DM + k)*64 + o]; },
      [&](int m,int n,float v){ T[m*LDF+n] = v; });
    __syncthreads();
    if(tid < 4*NODES){
      const int h=tid/NODES, i=tid%NODES;
      const float* a1 = p.gat_a1 + (L*4+h)*64;
      const float* a2 = p.gat_a2 + (L*4+h)*64;
      float x1=0.f, x2=0.f;
      for(int o=0;o<64;++o){ float w=T[i*LDF + h*64 + o]; x1=fmaf(w,a1[o],x1); x2=fmaf(w,a2[o],x2); }
      s_s1[h*NODES+i]=x1; s_s2[h*NODES+i]=x2;
    }
    __syncthreads();
    if(tid < 4*NODES){
      const int h=tid/NODES, i=tid%NODES;
      const unsigned msk = s_adj[i];
      float e[NODES]; float mx=NEGF;
      for(int j=0;j<NODES;++j){
        if((msk>>j)&1u){
          float v = s_s1[h*NODES+i] + s_s2[h*NODES+j];
          v = (v>0.f)? v : 0.2f*v;                      // LeakyReLU(0.2)
          e[j]=v; mx=fmaxf(mx,v);
        } else e[j]=NEGF;
      }
      float sum=0.f;
      for(int j=0;j<NODES;++j){ float pr=(e[j]<=NEGF*0.5f)?0.f:__expf(e[j]-mx); e[j]=pr; sum+=pr; }
      const float inv=1.f/sum;
      for(int j=0;j<NODES;++j) s_al[(h*NODES+i)*NODES + j] = e[j]*inv;
    }
    __syncthreads();
    for(int idx=tid; idx<NODES*DM; idx+=256){           // alpha @ Wh
      const int i=idx>>8, c=idx&255, h=c>>6;
      float s=0.f;
      for(int j=0;j<NODES;++j) s = fmaf(s_al[(h*NODES+i)*NODES+j], T[j*LDF+c], s);
      X16[i*LD16+c] = (_Float16)s;
    }
    __syncthreads();
    gemm16(X16, LD16, DM,                               // proj + residual
      [&](int k,int c){ return p.gat_pw[((size_t)L*DM + k)*DM + c]; },
      [&](int m,int n,float v){ T[m*LDF+n] = v + p.gat_pb[L*DM+n] + (float)H16[m*LD16+n]; });
    __syncthreads();
    lnorm(T, p.gat_g + L*DM, p.gat_b + L*DM, H16, nullptr);
    __syncthreads();
  }

  // ===== cross-attention x2: flash over 196 keys, double-buffered TDM staging =====
  for(int L=0; L<2; ++L){
    gemm16(H16, LD16, DM,
      [&](int k,int c){ return p.ca_qw[((size_t)L*DM+k)*DM + c]; },
      [&](int m,int n,float v){ Q16[m*LD16+n] = (_Float16)(v + p.ca_qb[L*DM+n]); });
    s_m[tid]=NEGF; s_l[tid]=0.f;
    v8f Ozero = {};
    v8f Oacc[4];
    #pragma unroll
    for(int q=0;q<4;++q) Oacc[q]=Ozero;
    const float* kwl = p.ca_kw + (size_t)L*DM*DM; const float* kbl = p.ca_kb + L*DM;
    const float* vwl = p.ca_vw + (size_t)L*DM*DM; const float* vbl = p.ca_vb + L*DM;
    const float* imgb = p.img + (size_t)b*NKEY*DM;

    // TDM issue: 2D tensor 196x256 f32, tile 32x256, LDS row pad 256->264 dwords.
    auto tdm_issue = [&](int kb0, unsigned ldsoff){
      const unsigned long long ga =
          (unsigned long long)(uintptr_t)(imgb + (size_t)kb0*DM);
      u32x4 g0 = { 1u,                                   // count=1 valid descriptor
                   ldsoff,                               // lds_addr
                   (unsigned)(ga & 0xffffffffull),       // global_addr[95:64]
                   (unsigned)(((ga>>32)&0x1ffffffull) | (2u<<30)) }; // addr hi | type=2
      u32x8 g1 = { (2u<<16) | (1u<<20) | (7u<<22) | (7u<<25), // dsz=4B, pad_en, 256dw/+8dw
                   (256u<<16),                           // tensor_dim0 = 256
                   (196u<<16),                           // tensor_dim1 = 196 (OOB rows -> 0)
                   (256u<<16),                           // tile_dim0 = 256
                   32u,                                  // tile_dim1 = 32
                   256u, 0u, 0u };                       // dim0_stride = 256
      asm volatile("tensor_load_to_lds %0, %1" :: "s"(g0), "s"(g1) : "memory");
    };

    if(wid==0){ tdm_issue(0, (unsigned)OFF_TA); __builtin_amdgcn_s_wait_tensorcnt(0); }
    __syncthreads();

    for(int kt=0; kt<7; ++kt){
      const int kb0 = kt*32;
      const float* cur = (kt&1)? TB : TA;
      if(wid==0 && kt<6)                                // async DMA of next tile
        tdm_issue(kb0+32, (kt&1)? (unsigned)OFF_TA : (unsigned)OFF_TB);
      if(kt < 5){                                       // L2 prefetch two tiles ahead
        const int nb = kb0 + 64;
        const int row = (tid*32) >> 8;
        if(nb + row < NKEY) __builtin_prefetch(imgb + (size_t)nb*DM + tid*32, 0, 0);
      }
      gemm_f32A(cur, LDF, DM,                           // K tile (row-major f16)
        [&](int k,int c){ return kwl[k*DM+c]; },
        [&](int m,int n,float v){ KV16[m*LD16+n]=(_Float16)(v + kbl[n]); });
      __syncthreads();
      for(int t=wid; t<32; t+=8){                       // S = Q K^T (hd=32 -> 1 wmma)
        const int h=t>>2, mi=t&1, nj=(t>>1)&1;
        v16h a  = frag_rowpair(Q16 + (size_t)(mi*16)*LD16 + h*32, LD16);
        v16h bb = frag_bt(KV16 + (size_t)(nj*16)*LD16 + h*32, LD16);
        v8f acc = {};
        acc = wmma16(a, bb, acc);
        frag_c_store(acc, [&](int m,int n,float v){
          S16[(h*32+mi*16+m)*LDS_S + nj*16+n] = (_Float16)(v*SCALE); });
      }
      __syncthreads();
      {                                                 // online softmax per (head,row)
        const int h=tid>>5, r=tid&31;
        if(r<NODES){
          const float mo=s_m[h*32+r];
          float sc[32]; float mx=mo;
          #pragma unroll
          for(int k=0;k<32;++k){
            const float s = (kb0+k<NKEY)? (float)S16[(h*32+r)*LDS_S + k] : NEGF;
            sc[k]=s; mx=fmaxf(mx,s);
          }
          const float cr=__expf(mo-mx);
          float lsum=s_l[h*32+r]*cr;
          #pragma unroll
          for(int k=0;k<32;++k){
            const float pr = (sc[k]<=NEGF*0.5f)? 0.f : __expf(sc[k]-mx);
            lsum += pr;
            S16[(h*32+r)*LDS_S + k] = (_Float16)pr;
          }
          s_m[h*32+r]=mx; s_l[h*32+r]=lsum; s_cor[h*32+r]=cr;
        } else s_cor[tid] = 1.f;
      }
      __syncthreads();
      gemm_f32A(cur, LDF, DM,                           // V tile stored TRANSPOSED [256][32]
        [&](int k,int c){ return vwl[k*DM+c]; },
        [&](int m,int n,float v){ KV16[(size_t)n*32 + m]=(_Float16)(v + vbl[n]); });
      __syncthreads();
      #pragma unroll
      for(int q=0;q<4;++q){                             // O = corr*O + P V
        const int t = wid + 8*q;
        const int mi=t&1, ni=t>>1, h=ni>>1;
        const int g=laneid()>>4;
        #pragma unroll
        for(int r8=0;r8<8;++r8) Oacc[q][r8] *= s_cor[h*32 + mi*16 + r8 + g*8];
        v16h a  = frag_rowpair(S16 + (size_t)(h*32+mi*16)*LDS_S, LDS_S);
        v16h bb = frag_bt(KV16 + (size_t)(ni*16)*32, 32);
        Oacc[q] = wmma16(a, bb, Oacc[q]);
      }
      if(wid==0 && kt<6) __builtin_amdgcn_s_wait_tensorcnt(0);  // next tile landed
      __syncthreads();
    }
    #pragma unroll
    for(int q=0;q<4;++q){                               // O / l -> f16
      const int t = wid + 8*q;
      const int mi=t&1, ni=t>>1, h=ni>>1;
      frag_c_store(Oacc[q], [&](int m,int n,float v){
        const int mm=mi*16+m;
        const float invl = 1.f / fmaxf(s_l[h*32+mm], 1e-30f);
        X16[mm*LD16 + ni*16+n] = (_Float16)(v * invl); });
    }
    __syncthreads();
    gemm16(X16, LD16, DM,                               // out-proj + residual
      [&](int k,int c){ return p.ca_ow[((size_t)L*DM+k)*DM + c]; },
      [&](int m,int n,float v){ T[m*LDF+n] = v + p.ca_ob[L*DM+n] + (float)H16[m*LD16+n]; });
    __syncthreads();
    lnorm(T, p.ca_g1 + L*DM, p.ca_b1 + L*DM, H16, nullptr);
    __syncthreads();
    ffn_block(p.ca_f1w + (size_t)L*DM*1024, p.ca_f1b + L*1024,
              p.ca_f2w + (size_t)L*1024*DM, p.ca_f2b + L*DM,
              1024, p.ca_g2 + L*DM, p.ca_b2 + L*DM, H16, Q16, T, nullptr);
  }

  // ================= transformer encoder x2 (self-attn, 21 tokens) =================
  for(int L=0; L<2; ++L){
    gemm16(H16, LD16, DM,
      [&](int k,int c){ return p.tr_qw[((size_t)L*DM+k)*DM+c]; },
      [&](int m,int n,float v){ Q16[m*LD16+n]=(_Float16)(v + p.tr_qb[L*DM+n]); });
    gemm16(H16, LD16, DM,
      [&](int k,int c){ return p.tr_kw[((size_t)L*DM+k)*DM+c]; },
      [&](int m,int n,float v){ X16[m*LD16+n]=(_Float16)(v + p.tr_kb[L*DM+n]); });
    gemm16(H16, LD16, DM,                               // V transposed [256][32]
      [&](int k,int c){ return p.tr_vw[((size_t)L*DM+k)*DM+c]; },
      [&](int m,int n,float v){ KV16[(size_t)n*32 + m]=(_Float16)(v + p.tr_vb[L*DM+n]); });
    __syncthreads();
    for(int t=wid; t<32; t+=8){
      const int h=t>>2, mi=t&1, nj=(t>>1)&1;
      v16h a  = frag_rowpair(Q16 + (size_t)(mi*16)*LD16 + h*32, LD16);
      v16h bb = frag_bt(X16 + (size_t)(nj*16)*LD16 + h*32, LD16);
      v8f acc={}; acc=wmma16(a,bb,acc);
      frag_c_store(acc,[&](int m,int n,float v){
        S16[(h*32+mi*16+m)*LDS_S + nj*16+n]=(_Float16)(v*SCALE); });
    }
    __syncthreads();
    {
      const int h=tid>>5, r=tid&31;
      if(r<NODES){
        float mx=NEGF;
        for(int j=0;j<NODES;++j) mx=fmaxf(mx,(float)S16[(h*32+r)*LDS_S+j]);
        float pr[NODES]; float sum=0.f;
        for(int j=0;j<NODES;++j){ const float e_=__expf((float)S16[(h*32+r)*LDS_S+j]-mx); pr[j]=e_; sum+=e_; }
        const float inv=1.f/sum;
        for(int j=0;j<NODES;++j) S16[(h*32+r)*LDS_S+j]=(_Float16)(pr[j]*inv);
        for(int j=NODES;j<32;++j) S16[(h*32+r)*LDS_S+j]=(_Float16)0.f;  // mask pad keys
      }
    }
    __syncthreads();
    #pragma unroll
    for(int q=0;q<4;++q){
      const int t=wid+8*q, mi=t&1, ni=t>>1, h=ni>>1;
      v16h a  = frag_rowpair(S16 + (size_t)(h*32+mi*16)*LDS_S, LDS_S);
      v16h bb = frag_bt(KV16 + (size_t)(ni*16)*32, 32);
      v8f acc={}; acc=wmma16(a,bb,acc);
      frag_c_store(acc,[&](int m,int n,float v){
        X16[(mi*16+m)*LD16 + ni*16+n]=(_Float16)v; });
    }
    __syncthreads();
    gemm16(X16, LD16, DM,
      [&](int k,int c){ return p.tr_ow[((size_t)L*DM+k)*DM+c]; },
      [&](int m,int n,float v){ T[m*LDF+n] = v + p.tr_ob[L*DM+n] + (float)H16[m*LD16+n]; });
    __syncthreads();
    lnorm(T, p.tr_g1+L*DM, p.tr_b1+L*DM, H16, nullptr);
    __syncthreads();
    ffn_block(p.tr_f1w + (size_t)L*DM*512, p.tr_f1b + L*512,
              p.tr_f2w + (size_t)L*512*DM, p.tr_f2b + L*DM,
              512, p.tr_g2+L*DM, p.tr_b2+L*DM, H16, Q16, T,
              (L==1)? (p.out + (size_t)b*NODES*DM) : nullptr);
  }
}

extern "C" void kernel_launch(void* const* d_in, const int* in_sizes, int n_in,
                              void* d_out, int out_size, void* d_ws, size_t ws_size,
                              hipStream_t stream){
  (void)n_in; (void)out_size; (void)d_ws; (void)ws_size;
  Params p;
  p.node_feats=(const float*)d_in[0];  p.img=(const float*)d_in[1];
  p.mlp_w1=(const float*)d_in[2];  p.mlp_b1=(const float*)d_in[3];
  p.mlp_w2=(const float*)d_in[4];  p.mlp_b2=(const float*)d_in[5];
  p.gat_W=(const float*)d_in[6];   p.gat_a1=(const float*)d_in[7];  p.gat_a2=(const float*)d_in[8];
  p.gat_pw=(const float*)d_in[9];  p.gat_pb=(const float*)d_in[10];
  p.gat_g=(const float*)d_in[11];  p.gat_b=(const float*)d_in[12];
  p.ca_qw=(const float*)d_in[13];  p.ca_qb=(const float*)d_in[14];
  p.ca_kw=(const float*)d_in[15];  p.ca_kb=(const float*)d_in[16];
  p.ca_vw=(const float*)d_in[17];  p.ca_vb=(const float*)d_in[18];
  p.ca_ow=(const float*)d_in[19];  p.ca_ob=(const float*)d_in[20];
  p.ca_g1=(const float*)d_in[21];  p.ca_b1=(const float*)d_in[22];
  p.ca_g2=(const float*)d_in[23];  p.ca_b2=(const float*)d_in[24];
  p.ca_f1w=(const float*)d_in[25]; p.ca_f1b=(const float*)d_in[26];
  p.ca_f2w=(const float*)d_in[27]; p.ca_f2b=(const float*)d_in[28];
  p.tr_qw=(const float*)d_in[29];  p.tr_qb=(const float*)d_in[30];
  p.tr_kw=(const float*)d_in[31];  p.tr_kb=(const float*)d_in[32];
  p.tr_vw=(const float*)d_in[33];  p.tr_vb=(const float*)d_in[34];
  p.tr_ow=(const float*)d_in[35];  p.tr_ob=(const float*)d_in[36];
  p.tr_g1=(const float*)d_in[37];  p.tr_b1=(const float*)d_in[38];
  p.tr_g2=(const float*)d_in[39];  p.tr_b2=(const float*)d_in[40];
  p.tr_f1w=(const float*)d_in[41]; p.tr_f1b=(const float*)d_in[42];
  p.tr_f2w=(const float*)d_in[43]; p.tr_f2b=(const float*)d_in[44];
  p.out=(float*)d_out;

  const int B = in_sizes[0] / (21*26);
  (void)hipFuncSetAttribute((const void*)hand_encoder_fused,
                            hipFuncAttributeMaxDynamicSharedMemorySize,
                            (int)SMEM_BYTES);
  hand_encoder_fused<<<dim3(B), dim3(256), SMEM_BYTES, stream>>>(p);
}
